// EdgeNeuralNetwork_81166291960074
// MI455X (gfx1250) — matrix operand order repaired and checked by reading, not compiled, not run on hardware
//
#include <hip/hip_runtime.h>

// ---------------------------------------------------------------------------
// MI455X (gfx1250, wave32) implementation.
// Factored algorithm: never materialize per-edge HxH matrices (268 MB, 17 GF).
//   m[b,j,u] = sum_{i,c} (adj*relu(edge@W1+b1))[b,j,(i,c)] * T[b,(i,c),u]
//              + sum_i adj[b,i,j]*Tb2[b,i,u]
//   T[b,i,c,u] = sum_v W2[c,u*32+v] * h[b,i,v]
// => ~4 GFLOP, ~34 MB f16 intermediates (fits 192 MB L2; ~1.5us at 23.3 TB/s).
// All GEMMs on V_WMMA_F32_16X16X32_F16, fp32 accumulate, fp32 hidden state.
// A-slab staging in the message GEMM uses GLOBAL_LOAD_ASYNC_TO_LDS_B128
// (ASYNCcnt) so the global->LDS copy bypasses VGPRs and overlaps with the
// register-based B transpose packing.
// ---------------------------------------------------------------------------

typedef __attribute__((ext_vector_type(16))) _Float16 v16h;
typedef __attribute__((ext_vector_type(8)))  _Float16 v8h;
typedef __attribute__((ext_vector_type(8)))  float    v8f;

#define KAR  4096              // N*FC : K of the message GEMM
#define BSTR 264               // padded k-stride of transposed LDS T slab

__device__ __forceinline__ v8f wmma16(const v16h a, const v16h b, const v8f c) {
  return __builtin_amdgcn_wmma_f32_16x16x32_f16(false, a, false, b, (short)0, c,
                                                false, false);
}
__device__ __forceinline__ v16h cat16(const v8h lo, const v8h hi) {
  return __builtin_shufflevector(lo, hi, 0, 1, 2, 3, 4, 5, 6, 7,
                                 8, 9, 10, 11, 12, 13, 14, 15);
}
__device__ __forceinline__ float sigmoidf_(float x) {
  return 1.0f / (1.0f + __expf(-x));
}

// Async global->LDS 16B copy (CDNA5 GLOBAL_LOAD_ASYNC_TO_LDS_B128, ASYNCcnt).
// Generic pointers to LDS carry the LDS byte offset in their low 32 bits
// (ISA flat addressing), which is what VDST wants.
__device__ __forceinline__ void async_b128(void* dst_lds, const void* src) {
  const unsigned lds = (unsigned)(unsigned long long)dst_lds;
  asm volatile("global_load_async_to_lds_b128 %0, %1, off"
               :: "v"(lds), "v"((unsigned long long)src)
               : "memory");
}
__device__ __forceinline__ void wait_async0() {
  asm volatile("s_wait_asynccnt 0x0" ::: "memory");
}

// ---------------------------------------------------------------------------
// Pre-swizzle W2 into k_T's B-fragment layout (contiguous k order, same as its A):
// W2B[nt][lane][e] (f16) with K = (lane<16?0:16)+e, col n = nt*16+(lane&15),
// value = W2T[k][n] where W2T[v][c*32+u] = W2[c][u*32+v].
// ---------------------------------------------------------------------------
__global__ void k_prep_w2(const float* __restrict__ W2, _Float16* __restrict__ W2B) {
  int tid  = blockIdx.x * 256 + threadIdx.x;      // 0 .. 131071
  int nt   = tid >> 9;
  int rem  = tid & 511;
  int lane = rem >> 4;
  int e    = rem & 15;
  int col  = nt * 16 + (lane & 15);
  int k    = ((lane < 16) ? 0 : 16) + e;          // v index (0..31)
  int c    = col >> 5;
  int u    = col & 31;
  W2B[tid] = (_Float16)W2[c * 1024 + u * 32 + k];
}

// ---------------------------------------------------------------------------
// Edge MLP stage-1 + relu + adjacency fold:
// AR[b][j][i*128+c] = adj[b,i,j] * relu(edge[b,i,j,:]@W1 + b1)[c]   (f16)
// GEMM M=65536 edges, K=16 (padded to 32), N=128. 1 wave = 1 M-tile (16 edges
// of fixed (b,i)), 8 N-tiles => 8 WMMAs / wave. W1 staged in LDS already in
// per-lane B-fragment layout so each fragment is one aligned 32B LDS read.
// ---------------------------------------------------------------------------
__global__ void k_edge_mlp(const float* __restrict__ edge,
                           const float* __restrict__ adj,
                           const float* __restrict__ W1,
                           const float* __restrict__ b1,
                           _Float16* __restrict__ AR) {
  __shared__ __align__(32) _Float16 sW1f[8 * 32 * 16];   // fragment-packed, 8 KB
  __shared__ float sb1[128];
  const int t = threadIdx.x;
#pragma unroll
  for (int r = 0; r < 16; ++r) {
    const int idx  = t + r * 256;                 // 0..4095
    const int nt   = idx >> 9;
    const int ln   = (idx >> 4) & 31;
    const int e    = idx & 15;
    const int col  = nt * 16 + (ln & 15);
    const float wv = W1[(((ln < 16) ? 0 : 8) + (e & 7)) * 128 + col]; // unconditional load
    sW1f[idx] = (_Float16)((e < 8) ? wv : 0.0f); // K>=16 rows are zero padding
  }
  if (t < 128) sb1[t] = b1[t];
  __syncthreads();

  const int w    = t >> 5;
  const int lane = t & 31;
  const int bi   = blockIdx.x * 4 + (w >> 1);     // (b*32 + i)
  const int b    = bi >> 5, i = bi & 31;
  const int j0   = (w & 1) * 16;

  // A fragment: 16 edge rows (K valid 0..15, 16..31 zero-padded)
  const int m  = lane & 15;
  const int kb = (lane < 16) ? 0 : 8;
  v16h a;
#pragma unroll
  for (int tt = 0; tt < 8; ++tt) {
    a[tt]     = (_Float16)edge[((long)bi * 32 + j0 + m) * 16 + kb + tt];
    a[8 + tt] = (_Float16)0.0f;                   // K >= 16 padding
  }
  const int rbase = (lane < 16) ? 0 : 8;
  float adjv[8];
#pragma unroll
  for (int r = 0; r < 8; ++r)
    adjv[r] = adj[(long)bi * 32 + j0 + rbase + r];

#pragma unroll
  for (int nt = 0; nt < 8; ++nt) {
    const int col = nt * 16 + (lane & 15);
    const v16h bb = *((const v16h*)(sW1f + (nt * 32 + lane) * 16)); // 32B aligned
    v8f c = {};
    c = wmma16(a, bb, c);
    const float bias = sb1[col];
#pragma unroll
    for (int r = 0; r < 8; ++r) {
      const int j = j0 + rbase + r;
      float v = c[r] + bias;
      v = v > 0.0f ? v : 0.0f;
      v *= adjv[r];
      AR[((long)(b * 32 + j) * 32 + i) * 128 + col] = (_Float16)v;
    }
  }
}

// ---------------------------------------------------------------------------
// h = x @ W_emb + b_emb ; msk[b,n] = (sum_f x > 0)
// ---------------------------------------------------------------------------
__global__ void k_embed(const float* __restrict__ x, const float* __restrict__ We,
                        const float* __restrict__ be, float* __restrict__ h,
                        float* __restrict__ msk) {
  const int tid = blockIdx.x * 256 + threadIdx.x; // 65536
  const int u   = tid & 31;
  const float* xr = x + (tid >> 5) * 32;
  float acc = be[u];
#pragma unroll
  for (int k = 0; k < 32; ++k) acc += xr[k] * We[k * 32 + u];
  h[tid] = acc;
  if (u == 0) {
    float s = 0.0f;
#pragma unroll
    for (int k = 0; k < 32; ++k) s += xr[k];
    msk[tid >> 5] = (s > 0.0f) ? 1.0f : 0.0f;
  }
}

// ---------------------------------------------------------------------------
// Per layer: h16 = f16(h);  Tb2[b,i,u] = sum_v b2[u*32+v]*h[b,i,v]
// ---------------------------------------------------------------------------
__global__ void k_hprep(const float* __restrict__ h, const float* __restrict__ b2,
                        _Float16* __restrict__ h16, float* __restrict__ Tb2) {
  const int tid = blockIdx.x * 256 + threadIdx.x; // 65536
  const int u   = tid & 31;
  h16[tid] = (_Float16)h[tid];
  const float* hr = h + (tid >> 5) * 32;
  float acc = 0.0f;
#pragma unroll
  for (int v = 0; v < 32; ++v) acc += b2[u * 32 + v] * hr[v];
  Tb2[tid] = acc;
}

// ---------------------------------------------------------------------------
// T GEMM: T[row=(b,i)][n=c*32+u] = sum_v h16[row][v] * W2T[v][n]
// M=2048, N=4096, K=32 -> exactly one WMMA per 16x16 tile. B fragment is one
// contiguous 32B global load thanks to the W2B pre-swizzle.
// ---------------------------------------------------------------------------
__global__ void k_T(const _Float16* __restrict__ h16,
                    const _Float16* __restrict__ W2B,
                    _Float16* __restrict__ T) {
  const int mt   = blockIdx.x >> 5;               // 0..127
  const int nb   = blockIdx.x & 31;
  const int w    = threadIdx.x >> 5;
  const int lane = threadIdx.x & 31;
  const int nt   = nb * 8 + w;                    // 0..255

  const int m  = lane & 15;
  const int kb = (lane < 16) ? 0 : 16;
  const _Float16* ar = h16 + (mt * 16 + m) * 32;
  const v8h alo = *((const v8h*)(ar + kb));
  const v8h ahi = *((const v8h*)(ar + kb + 8));
  const v16h a  = cat16(alo, ahi);

  const v16h bv = *((const v16h*)(W2B + ((long)nt * 32 + lane) * 16)); // 32B aligned
  v8f c = {};
  c = wmma16(a, bv, c);

  const int rbase = (lane < 16) ? 0 : 8;
  const int col   = nt * 16 + (lane & 15);
#pragma unroll
  for (int r = 0; r < 8; ++r)
    T[(long)(mt * 16 + rbase + r) * KAR + col] = (_Float16)c[r];
}

// ---------------------------------------------------------------------------
// Message GEMM per b:  m[b] = AR[b](32x4096) @ T[b](4096x32)  + adjT @ Tb2.
// K chunks of 256 staged through LDS:
//  - A slab + adj/Tb2 via GLOBAL_LOAD_ASYNC_TO_LDS_B128 (ASYNCcnt, no VGPR
//    roundtrip), drained with s_wait_asynccnt before the barrier;
//  - T slab stored TRANSPOSED ([u][k], stride 264 halves) via packed
//    ds_store_b32 so every B fragment is two conflict-free ds_load_b128
//    (lanes 0..15 hit banks 4*lane..4*lane+3 -> full 64-bank use).
// 8 WMMAs/chunk/wave; next chunk prefetched via global_prefetch_b8.
// Block = 128 threads (4 waves): wave -> (mt, ntile) quadrant of 32x32 output.
// ---------------------------------------------------------------------------
__global__ void k_msg(const _Float16* __restrict__ AR, const _Float16* __restrict__ T,
                      const float* __restrict__ adj, const float* __restrict__ Tb2,
                      float* __restrict__ mout) {
  __shared__ __align__(16) _Float16 sA[32 * 256];   // [j][k]          16 KB
  __shared__ __align__(16) _Float16 sBt[32 * BSTR]; // [u][k] padded  16.5 KB
  __shared__ __align__(16) float    sAdj[1024];
  __shared__ __align__(16) float    sT2[1024];
  const int b    = blockIdx.x;
  const int t    = threadIdx.x;                   // 0..127
  const int w    = t >> 5, lane = t & 31;
  const int mt   = w >> 1, ntl = w & 1;
  const long base = (long)b * (32 * KAR);

  // stage adj / Tb2 (4KB each) straight into LDS via async DMA
#pragma unroll
  for (int q = 0; q < 2; ++q) {
    const int ch = t + q * 128;                   // 0..255 16B chunks
    async_b128(&sAdj[ch * 4], adj + (long)b * 1024 + ch * 4);
    async_b128(&sT2[ch * 4],  Tb2 + (long)b * 1024 + ch * 4);
  }

  v8f acc = {};
  for (int kc = 0; kc < 16; ++kc) {
    const long k0 = (long)kc * 256;
    __syncthreads();
    // A slab: 32 rows x 256 k (rows strided by 4096 in AR[b]); async -> LDS
#pragma unroll
    for (int q = 0; q < 8; ++q) {
      const int idx = t + q * 128;                // 0..1023 16B chunks
      const int j = idx >> 5, kk = (idx & 31) * 8;
      async_b128(&sA[j * 256 + kk], AR + base + (long)j * KAR + k0 + kk);
    }
    // B slab transposed: T[b][k0+kk][u] -> sBt[u][kk]; needs registers for the
    // transpose, so regular coalesced uint4 reads + packed ds_store_b32.
#pragma unroll
    for (int q = 0; q < 4; ++q) {
      const int micro = t + q * 128;              // 0..511 (2k x 8u tiles)
      const int k  = (micro >> 2) * 2;            // even, 0..254
      const int u0 = (micro & 3) * 8;
      union { uint4 v; unsigned short s[8]; } r0, r1;
      r0.v = *((const uint4*)(T + base + (k0 + k) * 32 + u0));
      r1.v = *((const uint4*)(T + base + (k0 + k + 1) * 32 + u0));
#pragma unroll
      for (int uu = 0; uu < 8; ++uu) {
        const unsigned pair = (unsigned)r0.s[uu] | ((unsigned)r1.s[uu] << 16);
        *((unsigned*)(&sBt[(u0 + uu) * BSTR + k])) = pair;
      }
    }
    wait_async0();                                // drain ASYNCcnt before barrier
    __syncthreads();

    if (kc < 15)     // prefetch next T chunk -> global_prefetch_b8
      __builtin_prefetch((const void*)(T + base + (k0 + 256) * 32 + t * 64), 0, 1);

    const int mrow = mt * 16 + (lane & 15);
    const int ucol = ntl * 16 + (lane & 15);
#pragma unroll
    for (int kt = 0; kt < 8; ++kt) {
      const int kbA = kt * 32 + ((lane < 16) ? 0 : 8);
      const v8h alo = *((const v8h*)(&sA[mrow * 256 + kbA]));
      const v8h ahi = *((const v8h*)(&sA[mrow * 256 + kbA + 16]));
      const int kbB = kt * 32 + ((lane < 16) ? 0 : 8);
      const v8h blo = *((const v8h*)(&sBt[ucol * BSTR + kbB]));
      const v8h bhi = *((const v8h*)(&sBt[ucol * BSTR + kbB + 16]));
      acc = wmma16(cat16(alo, ahi), cat16(blo, bhi), acc);
    }
  }
  // epilogue: + sum_i adj[b,i,j]*Tb2[b,i,u], store fp32
#pragma unroll
  for (int r = 0; r < 8; ++r) {
    const int j = mt * 16 + ((lane < 16) ? r : 8 + r);
    const int u = ntl * 16 + (lane & 15);
    float v = acc[r];
    for (int i = 0; i < 32; ++i) v += sAdj[i * 32 + j] * sT2[i * 32 + u];
    mout[(long)b * 1024 + j * 32 + u] = v;
  }
}

// ---------------------------------------------------------------------------
// GRU cell (torch gate order r,z,n), masked update. One thread per (b,n,u).
// ---------------------------------------------------------------------------
__global__ void k_gru(const float* __restrict__ m, const float* __restrict__ h,
                      const float* __restrict__ Wi, const float* __restrict__ Wh,
                      const float* __restrict__ bi, const float* __restrict__ bh,
                      const float* __restrict__ msk, float* __restrict__ hout) {
  const int tid = blockIdx.x * 256 + threadIdx.x; // 65536
  const int bn  = tid >> 5, u = tid & 31;
  const float* mr = m + bn * 32;
  const float* hr = h + bn * 32;
  float gi0 = bi[u], gi1 = bi[32 + u], gi2 = bi[64 + u];
  float gh0 = bh[u], gh1 = bh[32 + u], gh2 = bh[64 + u];
#pragma unroll
  for (int k = 0; k < 32; ++k) {
    const float mv = mr[k], hv = hr[k];
    gi0 += mv * Wi[u * 32 + k];
    gi1 += mv * Wi[(32 + u) * 32 + k];
    gi2 += mv * Wi[(64 + u) * 32 + k];
    gh0 += hv * Wh[u * 32 + k];
    gh1 += hv * Wh[(32 + u) * 32 + k];
    gh2 += hv * Wh[(64 + u) * 32 + k];
  }
  const float r = sigmoidf_(gi0 + gh0);
  const float z = sigmoidf_(gi1 + gh1);
  const float n = tanhf(gi2 + r * gh2);
  hout[tid] = msk[bn] * ((1.0f - z) * n + z * hr[u]);
}

// ---------------------------------------------------------------------------
// Set2Set readout + output FC. Single 256-thread block, sequential 3 steps.
// ---------------------------------------------------------------------------
__global__ void k_s2s(const float* __restrict__ h, const float* __restrict__ msk,
                      const float* __restrict__ Wi, const float* __restrict__ Wh,
                      const float* __restrict__ lbi, const float* __restrict__ lbh,
                      const float* __restrict__ Wf, const float* __restrict__ bf,
                      float* __restrict__ out) {
  __shared__ float sq[64 * 32], sc[64 * 32], sqs[64 * 64], se[64 * 32], sa[64 * 32];
  const int t = threadIdx.x;
  for (int idx = t; idx < 64 * 32; idx += 256) { sq[idx] = 0.0f; sc[idx] = 0.0f; }
  for (int idx = t; idx < 64 * 64; idx += 256) sqs[idx] = 0.0f;
  __syncthreads();

  for (int step = 0; step < 3; ++step) {
    // LSTM cell: thread owns 8 (b,u) pairs; results staged in registers
    float qn[8], cn[8];
#pragma unroll
    for (int p = 0; p < 8; ++p) {
      const int idx = t + p * 256;
      const int b = idx >> 5, u = idx & 31;
      float g0 = lbi[u] + lbh[u];
      float g1 = lbi[32 + u] + lbh[32 + u];
      float g2 = lbi[64 + u] + lbh[64 + u];
      float g3 = lbi[96 + u] + lbh[96 + u];
      for (int k = 0; k < 64; ++k) {
        const float qs = sqs[b * 64 + k];
        g0 += qs * Wi[u * 64 + k];
        g1 += qs * Wi[(32 + u) * 64 + k];
        g2 += qs * Wi[(64 + u) * 64 + k];
        g3 += qs * Wi[(96 + u) * 64 + k];
      }
      for (int k = 0; k < 32; ++k) {
        const float qv = sq[b * 32 + k];
        g0 += qv * Wh[u * 32 + k];
        g1 += qv * Wh[(32 + u) * 32 + k];
        g2 += qv * Wh[(64 + u) * 32 + k];
        g3 += qv * Wh[(96 + u) * 32 + k];
      }
      const float cv = sigmoidf_(g1) * sc[idx] + sigmoidf_(g0) * tanhf(g2);
      cn[p] = cv;
      qn[p] = sigmoidf_(g3) * tanhf(cv);
    }
    __syncthreads();
#pragma unroll
    for (int p = 0; p < 8; ++p) { const int idx = t + p * 256; sc[idx] = cn[p]; sq[idx] = qn[p]; }
    __syncthreads();
    // attention logits e[b,n] = h[b,n,:].q[b,:]
#pragma unroll
    for (int p = 0; p < 8; ++p) {
      const int idx = t + p * 256;
      const int b = idx >> 5;
      const float* hrow = h + (long)idx * 32;
      float ev = 0.0f;
      for (int k = 0; k < 32; ++k) ev += hrow[k] * sq[b * 32 + k];
      se[idx] = (msk[idx] == 0.0f) ? -3.0e38f : ev;
    }
    __syncthreads();
    // softmax over n per graph
    if (t < 64) {
      float mx = -3.4e38f;
      for (int n = 0; n < 32; ++n) mx = fmaxf(mx, se[t * 32 + n]);
      float s = 0.0f;
      for (int n = 0; n < 32; ++n) { const float ex = __expf(se[t * 32 + n] - mx); sa[t * 32 + n] = ex; s += ex; }
      const float inv = 1.0f / s;
      for (int n = 0; n < 32; ++n) sa[t * 32 + n] *= inv;
    }
    __syncthreads();
    // r[b,u] = sum_n a*h*msk ; q_star = [q, r]
#pragma unroll
    for (int p = 0; p < 8; ++p) {
      const int idx = t + p * 256;
      const int b = idx >> 5, u = idx & 31;
      float rv = 0.0f;
      for (int n = 0; n < 32; ++n)
        rv += sa[b * 32 + n] * h[((long)b * 32 + n) * 32 + u] * msk[b * 32 + n];
      sqs[b * 64 + 32 + u] = rv;
      sqs[b * 64 + u]      = sq[idx];
    }
    __syncthreads();
  }
  if (t < 64) {
    float o = bf[0];
    for (int k = 0; k < 64; ++k) o += sqs[t * 64 + k] * Wf[k];
    out[t] = o;
  }
}

// ---------------------------------------------------------------------------
extern "C" void kernel_launch(void* const* d_in, const int* in_sizes, int n_in,
                              void* d_out, int out_size, void* d_ws, size_t ws_size,
                              hipStream_t stream) {
  (void)in_sizes; (void)n_in; (void)out_size; (void)ws_size;
  const float* x     = (const float*)d_in[0];
  const float* edge  = (const float*)d_in[1];
  const float* adj   = (const float*)d_in[2];
  const float* W_emb = (const float*)d_in[3];
  const float* b_emb = (const float*)d_in[4];
  const float* W1    = (const float*)d_in[5];
  const float* b1    = (const float*)d_in[6];
  const float* W2    = (const float*)d_in[7];
  const float* b2    = (const float*)d_in[8];
  const float* gWi   = (const float*)d_in[9];
  const float* gWh   = (const float*)d_in[10];
  const float* gbi   = (const float*)d_in[11];
  const float* gbh   = (const float*)d_in[12];
  const float* lWi   = (const float*)d_in[13];
  const float* lWh   = (const float*)d_in[14];
  const float* lbi   = (const float*)d_in[15];
  const float* lbh   = (const float*)d_in[16];
  const float* Wf    = (const float*)d_in[17];
  const float* bfp   = (const float*)d_in[18];
  float* out = (float*)d_out;

  char* ws = (char*)d_ws;
  size_t off = 0;
  auto alloc = [&](size_t bytes) -> void* {
    void* p = ws + off;
    off = (off + bytes + 255) & ~(size_t)255;
    return p;
  };
  _Float16* AR  = (_Float16*)alloc(8388608ull * 2);  // [B][j][i*128+c]
  _Float16* T   = (_Float16*)alloc(8388608ull * 2);  // [B*N rows][4096]
  _Float16* W2B = (_Float16*)alloc(131072ull * 2);   // swizzled B fragments
  _Float16* h16 = (_Float16*)alloc(65536ull * 2);
  float* hA   = (float*)alloc(65536ull * 4);
  float* hB   = (float*)alloc(65536ull * 4);
  float* Tb2  = (float*)alloc(65536ull * 4);
  float* mbuf = (float*)alloc(65536ull * 4);
  float* msk  = (float*)alloc(2048ull * 4);

  k_prep_w2<<<512, 256, 0, stream>>>(W2, W2B);
  k_edge_mlp<<<512, 256, 0, stream>>>(edge, adj, W1, b1, AR);
  k_embed<<<256, 256, 0, stream>>>(x, W_emb, b_emb, hA, msk);

  float* hcur = hA;
  float* hnxt = hB;
  for (int layer = 0; layer < 3; ++layer) {
    k_hprep<<<256, 256, 0, stream>>>(hcur, b2, h16, Tb2);
    k_T<<<4096, 256, 0, stream>>>(h16, W2B, T);
    k_msg<<<64, 128, 0, stream>>>(AR, T, adj, Tb2, mbuf);
    k_gru<<<256, 256, 0, stream>>>(mbuf, hcur, gWi, gWh, gbi, gbh, msk, hnxt);
    float* tmp = hcur; hcur = hnxt; hnxt = tmp;
  }
  k_s2s<<<1, 256, 0, stream>>>(hcur, msk, lWi, lWh, lbi, lbh, Wf, bfp, out);
}
